// GATNet_19782619366131
// MI455X (gfx1250) — compile-verified, matrix operand order checked by reading, not run on hardware
//
#include <hip/hip_runtime.h>

typedef __attribute__((ext_vector_type(16))) __bf16 v16bf;
typedef __attribute__((ext_vector_type(8)))  float  v8f;

#define N_NODES 20000
#define N_EDGES 320000

// ---------------------------------------------------------------------------
// fp32 -> bf16 cast
__global__ void cast_f32_bf16(const float* __restrict__ x, __bf16* __restrict__ y, int total) {
    int i = blockIdx.x * blockDim.x + threadIdx.x;
    if (i < total) y[i] = (__bf16)x[i];
}

// W [K x Nn] row-major fp32  ->  Wt [Nn x K] row-major bf16 (i.e. column-major B)
__global__ void wconv_kernel(const float* __restrict__ W, __bf16* __restrict__ Wt, int K, int Nn) {
    int i = blockIdx.x * blockDim.x + threadIdx.x;
    if (i >= K * Nn) return;
    int k = i / Nn, n = i - k * Nn;
    Wt[(size_t)n * K + k] = (__bf16)W[i];
}

// ---------------------------------------------------------------------------
// C[MxNn] = A[MxK](bf16,row-major) * B (given as Bt[NnxK] bf16 row-major)
// one wave -> 32x64 output tile: 2 (M) x 4 (N) blocking of 16x16 WMMA tiles,
// 8 WMMAs per K-step against 6 fragment loads.
// Requires M % 32 == 0, Nn % 64 == 0, K % 32 == 0 (true for all 9 GEMMs here).
__global__ void gemm_bf16_wmma(const __bf16* __restrict__ A, const __bf16* __restrict__ Bt,
                               float* __restrict__ C, int M, int Nn, int K) {
    const int lane = threadIdx.x & 31;
    const int wid  = (blockIdx.x * blockDim.x + threadIdx.x) >> 5;
    const int nt64 = Nn >> 6;
    const int mt   = wid / nt64;
    const int nt   = wid - mt * nt64;
    const int m0   = mt << 5, n0 = nt << 6;
    if (m0 >= M) return;
    const int half = lane >> 4;
    const int l    = lane & 15;
    const int kh   = half << 3;               // half-wave K offset (+8)

    v8f c0[4] = {{}, {}, {}, {}};             // M-tile 0, N-tiles 0..3
    v8f c1[4] = {{}, {}, {}, {}};             // M-tile 1, N-tiles 0..3

    const __bf16* a0p = A + (size_t)(m0 + l)      * K;
    const __bf16* a1p = A + (size_t)(m0 + 16 + l) * K;
    const __bf16* bp[4];
#pragma unroll
    for (int t = 0; t < 4; t++) bp[t] = Bt + (size_t)(n0 + 16 * t + l) * K;

    for (int k0 = 0; k0 < K; k0 += 32) {
        v16bf a0, a1, b[4];
#pragma unroll
        for (int i = 0; i < 16; i++) {
            int j = i >> 1;
            // ISA 16-bit A 16x32 layout: vgpr j -> K = (j>=4?16:0) + 2*(j&3) + (i&1) + 8*half
            int k = k0 + ((j & 4) << 2) + ((j & 3) << 1) + (i & 1) + kh;
            a0[i]   = a0p[k];
            a1[i]   = a1p[k];
            b[0][i] = bp[0][k];
            b[1][i] = bp[1][k];
            b[2][i] = bp[2][k];
            b[3][i] = bp[3][k];
        }
#pragma unroll
        for (int t = 0; t < 4; t++) {
            c0[t] = __builtin_amdgcn_wmma_f32_16x16x32_bf16(false, a0, false, b[t], (short)0, c0[t], false, false);
            c1[t] = __builtin_amdgcn_wmma_f32_16x16x32_bf16(false, a1, false, b[t], (short)0, c1[t], false, false);
        }
    }
    // C/D layout: vgpr r, lanes 0-15 -> M = r (+8*half via lanes 16-31), N = lane%16
#pragma unroll
    for (int r = 0; r < 8; r++) {
        int mr0 = m0 + kh + r;
        int mr1 = mr0 + 16;
#pragma unroll
        for (int t = 0; t < 4; t++) {
            C[(size_t)mr0 * Nn + n0 + 16 * t + l] = c0[t][r];
            C[(size_t)mr1 * Nn + n0 + 16 * t + l] = c1[t][r];
        }
    }
}

// ---------------------------------------------------------------------------
// edge pass 1: e = leaky_relu(xl[src]+xr[dst], 0.2) . att ; segment-max via ordered-uint atomicMax
__global__ void edge_scores(const float* __restrict__ xl, const float* __restrict__ xr,
                            const float* __restrict__ att, const int* __restrict__ src,
                            const int* __restrict__ dst, float* __restrict__ e,
                            unsigned* __restrict__ mord, int E, int D) {
    int wid  = (blockIdx.x * blockDim.x + threadIdx.x) >> 5;
    int lane = threadIdx.x & 31;
    if (wid >= E) return;
    int s = src[wid], d = dst[wid];
    const float* pl = xl + (size_t)s * D;
    const float* pr = xr + (size_t)d * D;
    float acc = 0.f;
    for (int j = lane; j < D; j += 32) {
        float v = pl[j] + pr[j];
        v = v > 0.f ? v : 0.2f * v;
        acc = fmaf(v, att[j], acc);
    }
#pragma unroll
    for (int off = 16; off > 0; off >>= 1) acc += __shfl_xor(acc, off, 32);
    if (lane == 0) {
        e[wid] = acc;
        unsigned u = __float_as_uint(acc);
        u = (u & 0x80000000u) ? ~u : (u | 0x80000000u);   // monotone float -> uint
        atomicMax(mord + d, u);
    }
}

// edge pass 2: ex = exp(e - m[dst]); s[dst] += ex   (in-place on e)
__global__ void edge_exp(float* __restrict__ e, const int* __restrict__ dst,
                         const unsigned* __restrict__ mord, float* __restrict__ ssum, int E) {
    int i = blockIdx.x * blockDim.x + threadIdx.x;
    if (i >= E) return;
    int d = dst[i];
    unsigned o = mord[d];
    unsigned u = (o & 0x80000000u) ? (o ^ 0x80000000u) : ~o;
    float m = __uint_as_float(u);
    float v = __expf(e[i] - m);
    e[i] = v;
    atomicAdd(ssum + d, v);
}

// edge pass 3: agg[dst] += (ex / (s[dst]+1e-16)) * xl[src]
__global__ void edge_aggregate(const float* __restrict__ ex, const float* __restrict__ ssum,
                               const float* __restrict__ xl, const int* __restrict__ src,
                               const int* __restrict__ dst, float* __restrict__ agg, int E, int D) {
    int wid  = (blockIdx.x * blockDim.x + threadIdx.x) >> 5;
    int lane = threadIdx.x & 31;
    if (wid >= E) return;
    int s = src[wid], d = dst[wid];
    float alpha = ex[wid] / (ssum[d] + 1e-16f);
    const float* pl = xl + (size_t)s * D;
    float*       pa = agg + (size_t)d * D;
    for (int j = lane; j < D; j += 32) atomicAdd(pa + j, alpha * pl[j]);
}

// combine (layers 1,2): h = relu(agg + bc + sk + bs) -> bf16 for next layer GEMMs
__global__ void combine_relu_bf16(const float* __restrict__ agg, const float* __restrict__ sk,
                                  const float* __restrict__ bc, const float* __restrict__ bs,
                                  __bf16* __restrict__ hb, int total, int Dmask) {
    int i = blockIdx.x * blockDim.x + threadIdx.x;
    if (i >= total) return;
    int j = i & Dmask;
    float v = agg[i] + bc[j] + sk[i] + bs[j];
    hb[i] = (__bf16)fmaxf(v, 0.f);
}

// combine (layer 3): fp32 output, no relu
__global__ void combine_final(const float* __restrict__ agg, const float* __restrict__ sk,
                              const float* __restrict__ bc, const float* __restrict__ bs,
                              float* __restrict__ out, int total, int Dmask) {
    int i = blockIdx.x * blockDim.x + threadIdx.x;
    if (i >= total) return;
    int j = i & Dmask;
    out[i] = agg[i] + bc[j] + sk[i] + bs[j];
}

// ---------------------------------------------------------------------------
static inline int cdiv(long long a, long long b) { return (int)((a + b - 1) / b); }

extern "C" void kernel_launch(void* const* d_in, const int* in_sizes, int n_in,
                              void* d_out, int out_size, void* d_ws, size_t ws_size,
                              hipStream_t stream) {
    (void)in_sizes; (void)n_in; (void)out_size; (void)ws_size;
    const int N = N_NODES, E = N_EDGES;
    const int DMAX = 512;
    const int dims_in[3]  = {256, 512, 256};
    const int dims_out[3] = {512, 256, 128};

    const float* x   = (const float*)d_in[0];
    const int*   src = (const int*)d_in[1];
    const int*   dst = src + E;

    // workspace carve-up (~147 MB)
    char* ws = (char*)d_ws;
    size_t off = 0;
    auto carve = [&](size_t bytes) { char* p = ws + off; off = (off + bytes + 255) & ~(size_t)255; return p; };
    __bf16*   hb   = (__bf16*)  carve((size_t)N * DMAX * 2);  // bf16 activations
    float*    xl   = (float*)   carve((size_t)N * DMAX * 4);
    float*    xr   = (float*)   carve((size_t)N * DMAX * 4);  // reused as agg after pass 1
    float*    sk   = (float*)   carve((size_t)N * DMAX * 4);
    float*    ebuf = (float*)   carve((size_t)E * 4);         // e, then ex (in place)
    unsigned* mord = (unsigned*)carve((size_t)N * 4);
    float*    ssum = (float*)   carve((size_t)N * 4);
    __bf16*   wtl  = (__bf16*)  carve((size_t)512 * 512 * 2);
    __bf16*   wtr  = (__bf16*)  carve((size_t)512 * 512 * 2);
    __bf16*   wts  = (__bf16*)  carve((size_t)512 * 512 * 2);

    const int TB = 256;

    // cast input features to bf16
    {
        int total = N * dims_in[0];
        cast_f32_bf16<<<cdiv(total, TB), TB, 0, stream>>>(x, hb, total);
    }

    for (int li = 0; li < 3; li++) {
        const int Din = dims_in[li], Dout = dims_out[li];
        const int base = 2 + 6 * li;
        const float* Wl  = (const float*)d_in[base + 0];
        const float* Wr  = (const float*)d_in[base + 1];
        const float* att = (const float*)d_in[base + 2];
        const float* bc  = (const float*)d_in[base + 3];
        const float* Ws  = (const float*)d_in[base + 4];
        const float* bs  = (const float*)d_in[base + 5];

        // weight -> transposed bf16
        int wtotal = Din * Dout;
        wconv_kernel<<<cdiv(wtotal, TB), TB, 0, stream>>>(Wl, wtl, Din, Dout);
        wconv_kernel<<<cdiv(wtotal, TB), TB, 0, stream>>>(Wr, wtr, Din, Dout);
        wconv_kernel<<<cdiv(wtotal, TB), TB, 0, stream>>>(Ws, wts, Din, Dout);

        // three GEMMs: xl, xr, skip  (wave = 32x64 tile)
        long long waves = (long long)(N / 32) * (Dout / 64);
        int gblocks = cdiv(waves * 32, TB);
        gemm_bf16_wmma<<<gblocks, TB, 0, stream>>>(hb, wtl, xl, N, Dout, Din);
        gemm_bf16_wmma<<<gblocks, TB, 0, stream>>>(hb, wtr, xr, N, Dout, Din);
        gemm_bf16_wmma<<<gblocks, TB, 0, stream>>>(hb, wts, sk, N, Dout, Din);

        // segment softmax + aggregation
        hipMemsetAsync(mord, 0, (size_t)N * 4, stream);  // ordered key 0 == below -inf
        hipMemsetAsync(ssum, 0, (size_t)N * 4, stream);
        edge_scores<<<cdiv((long long)E * 32, TB), TB, 0, stream>>>(xl, xr, att, src, dst, ebuf, mord, E, Dout);
        edge_exp<<<cdiv(E, TB), TB, 0, stream>>>(ebuf, dst, mord, ssum, E);
        hipMemsetAsync(xr, 0, (size_t)N * Dout * 4, stream);  // xr becomes agg
        edge_aggregate<<<cdiv((long long)E * 32, TB), TB, 0, stream>>>(ebuf, ssum, xl, src, dst, xr, E, Dout);

        // combine skip + attention output + biases
        int total = N * Dout;
        if (li < 2) {
            combine_relu_bf16<<<cdiv(total, TB), TB, 0, stream>>>(xr, sk, bc, bs, hb, total, Dout - 1);
        } else {
            combine_final<<<cdiv(total, TB), TB, 0, stream>>>(xr, sk, bc, bs, (float*)d_out, total, Dout - 1);
        }
    }
}
